// NormalizedReLU2HeadAssociativeLM_34368328303123
// MI455X (gfx1250) — matrix operand order repaired and checked by reading, compiled, not run on hardware
//
#include <hip/hip_runtime.h>
#include <hip/hip_bf16.h>
#include <cstdint>

typedef __attribute__((ext_vector_type(16))) __bf16 v16bf;
typedef __attribute__((ext_vector_type(8)))  __bf16 v8bf;
typedef __attribute__((ext_vector_type(8)))  float  v8f;

namespace {
constexpr int Bb = 2, Ss = 1024, Ee = 512, Hh = 1024, Mm = 128, Vv = 32000;
constexpr int LDP = 72;   // LDS row pitch in halfs: 64 data + 8 pad (bank spread)
}

// ---------------------------------------------------------------------------
// bf16 WMMA GEMM with async global->LDS double-buffered staging (CDNA5 path):
//   out[m,n] = sum_k A[m,k] * W[n,k] + bias[n]   (A: MdxKd, W: NdxKd, row-major)
// Block = 256 threads = 8 waves; block tile 64(M) x 128(N) x 64(K).
// Tiles staged with GLOBAL_LOAD_ASYNC_TO_LDS_B128 (ASYNCcnt), double buffered:
// issue tile k+1, s_wait_asynccnt 6 (own wave's older 6 complete), barrier,
// compute 2x 32-K WMMA steps from LDS (ds_load_b128 fragments), barrier.
// Fragment layouts per CDNA5 ISA tables:
//   A 16x32 : lane%16 = M row; lane/16 selects K-octets {0-7,16-23}/{8-15,24-31}
//   B 32x16 : lane%16 = N col; lane/16 selects K-octets {0-7,16-23}/{8-15,24-31}
//   D 16x16 : VGPR r -> M = r + 8*(lane/16); N = lane%16
// ep==1 applies relu^2. out_bf (optional) gets a bf16 copy for the next GEMM.
// ---------------------------------------------------------------------------
__global__ __launch_bounds__(256) void wmma_gemm_bias(
    const __bf16* __restrict__ A, const __bf16* __restrict__ W,
    const float* __restrict__ bias, float* __restrict__ out,
    __bf16* __restrict__ out_bf, int Md, int Nd, int Kd, int ep)
{
  __shared__ __bf16 sA[2][64 * LDP];
  __shared__ __bf16 sB[2][128 * LDP];

  const int tid  = threadIdx.x;
  const int lane = tid & 31;
  const int wave = tid >> 5;
  const int l    = lane & 15;
  const int hi   = lane >> 4;
  const int wm   = wave >> 2;           // 0..1 : 32-row strip in M
  const int wn   = wave & 3;            // 0..3 : 32-col strip in N
  const int bm   = blockIdx.y * 64;
  const int bn   = blockIdx.x * 128;

  // Stage one 64-K tile of A (64 rows) and W (128 rows) into LDS buffer `buf`.
  // 16B per async op; 2 ops (A) + 4 ops (W) per thread -> asynccnt += 6 / wave.
  auto issue_tile = [&](int kt, int buf) {
    const int k0 = kt * 64;
#pragma unroll
    for (int it = 0; it < 2; ++it) {               // A: 64 rows x 8 segs
      const int idx = tid + it * 256;
      const int r = idx >> 3, sg = idx & 7;
      const __bf16* gp = A + (size_t)(bm + r) * Kd + k0 + sg * 8;
      unsigned lo = (unsigned)(uintptr_t)&sA[buf][r * LDP + sg * 8];
      unsigned long long ga = (unsigned long long)(uintptr_t)gp;
      asm volatile("global_load_async_to_lds_b128 %0, %1, off"
                   :: "v"(lo), "v"(ga) : "memory");
    }
#pragma unroll
    for (int it = 0; it < 4; ++it) {               // W: 128 rows x 8 segs
      const int idx = tid + it * 256;
      const int r = idx >> 3, sg = idx & 7;
      const __bf16* gp = W + (size_t)(bn + r) * Kd + k0 + sg * 8;
      unsigned lo = (unsigned)(uintptr_t)&sB[buf][r * LDP + sg * 8];
      unsigned long long ga = (unsigned long long)(uintptr_t)gp;
      asm volatile("global_load_async_to_lds_b128 %0, %1, off"
                   :: "v"(lo), "v"(ga) : "memory");
    }
  };

  v8f acc[2][2] = {};
  const int KTn = Kd >> 6;
  issue_tile(0, 0);

  for (int kt = 0; kt < KTn; ++kt) {
    const int buf = kt & 1;
    if (kt + 1 < KTn) {
      issue_tile(kt + 1, buf ^ 1);                 // overlap DMA with compute
      asm volatile("s_wait_asynccnt 0x6" ::: "memory");  // tile kt landed
    } else {
      asm volatile("s_wait_asynccnt 0x0" ::: "memory");
    }
    __syncthreads();                               // all waves' tiles visible

#pragma unroll
    for (int ks = 0; ks < 2; ++ks) {               // two 32-K WMMA steps
      const int kk = ks * 32;
      v16bf af[2], bfr[2];
#pragma unroll
      for (int i = 0; i < 2; ++i) {
        const __bf16* ap = &sA[buf][(wm * 32 + i * 16 + l) * LDP + kk + hi * 8];
        v8bf alo = *(const v8bf*)ap;               // ds_load_b128
        v8bf ahi = *(const v8bf*)(ap + 16);
        af[i] = __builtin_shufflevector(alo, ahi,
                0,1,2,3,4,5,6,7,8,9,10,11,12,13,14,15);
        const __bf16* bp = &sB[buf][(wn * 32 + i * 16 + l) * LDP + kk + hi * 8];
        v8bf blo = *(const v8bf*)bp;
        v8bf bhi = *(const v8bf*)(bp + 16);
        bfr[i] = __builtin_shufflevector(blo, bhi,
                0,1,2,3,4,5,6,7,8,9,10,11,12,13,14,15);
      }
#pragma unroll
      for (int i = 0; i < 2; ++i)
#pragma unroll
        for (int j = 0; j < 2; ++j)
          acc[i][j] = __builtin_amdgcn_wmma_f32_16x16x32_bf16(
              false, af[i], false, bfr[j], (short)0, acc[i][j], false, false);
    }
    __syncthreads();         // buffer re-targeted by next iteration's DMA
  }

#pragma unroll
  for (int i = 0; i < 2; ++i) {
#pragma unroll
    for (int j = 0; j < 2; ++j) {
      const int n  = bn + wn * 32 + j * 16 + l;
      const float bv = bias[n];
#pragma unroll
      for (int r = 0; r < 8; ++r) {
        const int m = bm + wm * 32 + i * 16 + r + 8 * hi;
        float v = acc[i][j][r] + bv;
        if (ep == 1) { v = fmaxf(v, 0.f); v *= v; }
        out[(size_t)m * Nd + n] = v;
        if (out_bf) out_bf[(size_t)m * Nd + n] = (__bf16)v;
      }
    }
  }
}

// ---------------------------------------------------------------------------
// Persistent GRU scan. 64 workgroups, each owns 16 of the 1024 hidden units
// (-> the 3 gate rows j, j+H, j+2H of w_hh). h double-buffered in global
// memory; one monotonic-counter grid barrier per step. w_hh stays f32 and
// L2-resident (12.6 MB).
// ---------------------------------------------------------------------------
__global__ __launch_bounds__(256) void gru_scan(
    const float* __restrict__ xg, const float* __restrict__ w_hh,
    const float* __restrict__ b_hh, float* __restrict__ states,
    float* __restrict__ hbuf, unsigned* __restrict__ bar)
{
  const int tid = threadIdx.x, lane = tid & 31, wave = tid >> 5;
  const int j0 = blockIdx.x * 16;
  const unsigned nwg = gridDim.x;
  __shared__ float sh_h[2 * Hh];
  __shared__ float sh_hg[96];

  for (int s = 0; s < Ss; ++s) {
    const float* hin  = hbuf + (s & 1) * 2 * Hh;
    float*       hout = hbuf + ((s + 1) & 1) * 2 * Hh;
    for (int i = tid; i < 2 * Hh; i += 256) sh_h[i] = hin[i];
    __syncthreads();

    // 96 dot products (2 batches x 16 j x 3 gates), 12 per wave.
    for (int d = wave * 12; d < wave * 12 + 12; ++d) {
      const int b = d / 48, rem = d % 48, jj = rem / 3, g = rem % 3;
      const int row = g * Hh + j0 + jj;
      const float* wr = w_hh + (size_t)row * Hh;
      const float* hr = sh_h + b * Hh;
      float sum = 0.f;
      for (int kk = lane; kk < Hh; kk += 32) sum += hr[kk] * wr[kk];
      for (int o = 16; o; o >>= 1) sum += __shfl_down(sum, o, 32);
      if (lane == 0) sh_hg[d] = sum + b_hh[row];
    }
    __syncthreads();

    if (tid < 32) {
      const int b = tid >> 4, jj = tid & 15, j = j0 + jj;
      const float* xrow = xg + (size_t)(b * Ss + s) * (3 * Hh);
      const float hgr = sh_hg[b * 48 + jj * 3 + 0];
      const float hgz = sh_hg[b * 48 + jj * 3 + 1];
      const float hgn = sh_hg[b * 48 + jj * 3 + 2];
      const float r  = 1.f / (1.f + __expf(-(xrow[j] + hgr)));
      const float z  = 1.f / (1.f + __expf(-(xrow[Hh + j] + hgz)));
      const float nn = tanhf(xrow[2 * Hh + j] + r * hgn);
      const float hn = (1.f - z) * nn + z * sh_h[b * Hh + j];
      hout[b * Hh + j] = hn;
      states[((size_t)b * Ss + s) * Hh + j] = hn;
    }
    __threadfence();
    __syncthreads();
    if (tid == 0) {
      atomicAdd(bar, 1u);
      const unsigned target = (unsigned)(s + 1) * nwg;
      while (atomicAdd(bar, 0u) < target) __builtin_amdgcn_s_sleep(2);
    }
    __syncthreads();
  }
}

// ---------------------------------------------------------------------------
// LayerNorm + gate: one block per (b,s) row. Writes ns (bf16), states (bf16)
// and gate[row] = sigmoid(states_row . wg + bg) in one pass.
// ---------------------------------------------------------------------------
__global__ __launch_bounds__(256) void ln_gate(
    const float* __restrict__ states, const float* __restrict__ ln_g,
    const float* __restrict__ ln_b, const float* __restrict__ wg,
    const float* __restrict__ bg, __bf16* __restrict__ ns_bf,
    __bf16* __restrict__ st_bf, float* __restrict__ gate)
{
  const int row = blockIdx.x;
  const int tid = threadIdx.x, lane = tid & 31, wave = tid >> 5;
  const float* x = states + (size_t)row * Hh;
  float s1 = 0.f, s2 = 0.f, s3 = 0.f;
  for (int i = tid; i < Hh; i += 256) {
    const float v = x[i];
    s1 += v; s2 += v * v; s3 += v * wg[i];
  }
  for (int o = 16; o; o >>= 1) {
    s1 += __shfl_down(s1, o, 32);
    s2 += __shfl_down(s2, o, 32);
    s3 += __shfl_down(s3, o, 32);
  }
  __shared__ float r1[8], r2[8], r3[8];
  __shared__ float mu_s, rstd_s, dot_s;
  if (lane == 0) { r1[wave] = s1; r2[wave] = s2; r3[wave] = s3; }
  __syncthreads();
  if (tid == 0) {
    float a = 0.f, b2 = 0.f, c = 0.f;
    for (int w = 0; w < 8; ++w) { a += r1[w]; b2 += r2[w]; c += r3[w]; }
    const float mu = a / Hh;
    mu_s   = mu;
    rstd_s = rsqrtf(b2 / Hh - mu * mu + 1e-5f);
    dot_s  = c;
  }
  __syncthreads();
  const float mu = mu_s, rstd = rstd_s;
  for (int i = tid; i < Hh; i += 256) {
    const float v = x[i];
    ns_bf[(size_t)row * Hh + i] = (__bf16)((v - mu) * rstd * ln_g[i] + ln_b[i]);
    st_bf[(size_t)row * Hh + i] = (__bf16)v;
  }
  if (tid == 0) gate[row] = 1.f / (1.f + __expf(-(dot_s + bg[0])));
}

// ---------------------------------------------------------------------------
// Causal attention row + scatter-add into logits. One block per (b,i).
// Strict lower-triangular mask; softmax over j<i renormalized; duplicate
// token ids handled with atomicAdd (matches jnp .at[].add semantics).
// ---------------------------------------------------------------------------
__global__ __launch_bounds__(256) void attn_scatter(
    const float* __restrict__ q, const float* __restrict__ k,
    const float* __restrict__ gate, const int* __restrict__ ids,
    const float* __restrict__ mem_scale, float* __restrict__ out)
{
  const int b = blockIdx.x / Ss, i = blockIdx.x % Ss;
  if (i == 0) return;   // fully masked row contributes nothing
  const int tid = threadIdx.x, lane = tid & 31, wave = tid >> 5;
  __shared__ float qsh[Mm];
  __shared__ float sc[Ss];
  __shared__ float red[8];
  __shared__ float bc;
  if (tid < Mm) qsh[tid] = q[((size_t)b * Ss + i) * Mm + tid];
  __syncthreads();

  const float scale = rsqrtf((float)Mm);
  float lmax = -3.4e38f;
  for (int j = tid; j < i; j += 256) {
    const float4* kr = (const float4*)(k + ((size_t)b * Ss + j) * Mm);
    float dot = 0.f;
#pragma unroll
    for (int m = 0; m < Mm / 4; ++m) {
      const float4 kv = kr[m];
      dot += qsh[4 * m + 0] * kv.x + qsh[4 * m + 1] * kv.y +
             qsh[4 * m + 2] * kv.z + qsh[4 * m + 3] * kv.w;
    }
    dot *= scale;
    sc[j] = dot;
    lmax = fmaxf(lmax, dot);
  }
  for (int o = 16; o; o >>= 1) lmax = fmaxf(lmax, __shfl_down(lmax, o, 32));
  if (lane == 0) red[wave] = lmax;
  __syncthreads();
  if (tid == 0) {
    float m = red[0];
    for (int w = 1; w < 8; ++w) m = fmaxf(m, red[w]);
    bc = m;
  }
  __syncthreads();
  const float mx = bc;

  float lsum = 0.f;
  for (int j = tid; j < i; j += 256) {
    const float p = __expf(sc[j] - mx);
    sc[j] = p;
    lsum += p;
  }
  for (int o = 16; o; o >>= 1) lsum += __shfl_down(lsum, o, 32);
  __syncthreads();
  if (lane == 0) red[wave] = lsum;
  __syncthreads();
  if (tid == 0) {
    float sm = 0.f;
    for (int w = 0; w < 8; ++w) sm += red[w];
    bc = 1.f / fmaxf(sm, 1e-6f);
  }
  __syncthreads();

  const float coef = bc * gate[b * Ss + i] * mem_scale[0];
  float* orow = out + ((size_t)b * Ss + i) * Vv;
  for (int j = tid; j < i; j += 256)
    atomicAdd(orow + ids[b * Ss + j], sc[j] * coef);
}

// --------------------------- small helpers ---------------------------------
__global__ void cast_bf16(const float* __restrict__ in, __bf16* __restrict__ o,
                          long n) {
  long i  = (long)blockIdx.x * blockDim.x + threadIdx.x;
  long st = (long)gridDim.x * blockDim.x;
  for (; i < n; i += st) o[i] = (__bf16)in[i];
}

__global__ void gather_embed(const int* __restrict__ ids,
                             const float* __restrict__ emb,
                             __bf16* __restrict__ xbf) {
  const int row = blockIdx.x;
  const long t = ids[row];
  const float* src = emb + (size_t)t * Ee;
  __bf16* dst = xbf + (size_t)row * Ee;
  for (int e = threadIdx.x; e < Ee; e += blockDim.x) dst[e] = (__bf16)src[e];
}

__global__ void zero_f32(float* p, int n) {
  int i = blockIdx.x * blockDim.x + threadIdx.x;
  if (i < n) p[i] = 0.f;
}
__global__ void zero_u32(unsigned* p, int n) {
  int i = blockIdx.x * blockDim.x + threadIdx.x;
  if (i < n) p[i] = 0u;
}

// ---------------------------------------------------------------------------
extern "C" void kernel_launch(void* const* d_in, const int* in_sizes, int n_in,
                              void* d_out, int out_size, void* d_ws,
                              size_t ws_size, hipStream_t stream)
{
  const int*   ids  = (const int*)  d_in[0];
  const float* emb  = (const float*)d_in[1];
  const float* w_ih = (const float*)d_in[2];
  const float* b_ih = (const float*)d_in[3];
  const float* w_hh = (const float*)d_in[4];
  const float* b_hh = (const float*)d_in[5];
  const float* ln_g = (const float*)d_in[6];
  const float* ln_b = (const float*)d_in[7];
  const float* wq   = (const float*)d_in[8];
  const float* bq   = (const float*)d_in[9];
  const float* wk   = (const float*)d_in[10];
  const float* bk   = (const float*)d_in[11];
  const float* wg   = (const float*)d_in[12];
  const float* bg   = (const float*)d_in[13];
  const float* w_fc = (const float*)d_in[14];
  const float* b_fc = (const float*)d_in[15];
  const float* w_pr = (const float*)d_in[16];
  const float* b_pr = (const float*)d_in[17];
  const float* outb = (const float*)d_in[18];
  const float* msc  = (const float*)d_in[19];
  float* out = (float*)d_out;
  (void)in_sizes; (void)n_in; (void)out_size; (void)ws_size;

  char* base = (char*)d_ws;
  size_t off = 0;
  auto alloc = [&](size_t bytes) {
    char* p = base + off;
    off = (off + bytes + 255) & ~(size_t)255;
    return p;
  };
  const int R = Bb * Ss;  // 2048 rows
  __bf16* x_bf   = (__bf16*)alloc((size_t)R * Ee * 2);
  float*  xg     = (float*) alloc((size_t)R * 3 * Hh * 4);
  float*  states = (float*) alloc((size_t)R * Hh * 4);
  __bf16* st_bf  = (__bf16*)alloc((size_t)R * Hh * 2);
  __bf16* ns_bf  = (__bf16*)alloc((size_t)R * Hh * 2);
  float*  hf     = (float*) alloc((size_t)R * 4 * Ee * 4);
  __bf16* hf_bf  = (__bf16*)alloc((size_t)R * 4 * Ee * 2);
  float*  bs     = (float*) alloc((size_t)R * Ee * 4);
  __bf16* bs_bf  = (__bf16*)alloc((size_t)R * Ee * 2);
  float*  qb     = (float*) alloc((size_t)R * Mm * 4);
  float*  kb     = (float*) alloc((size_t)R * Mm * 4);
  float*  gate   = (float*) alloc((size_t)R * 4);
  float*  hbuf   = (float*) alloc((size_t)2 * Bb * Hh * 4);
  unsigned* bar  = (unsigned*)alloc(256);
  __bf16* wih_bf = (__bf16*)alloc((size_t)3 * Hh * Ee * 2);
  __bf16* wfc_bf = (__bf16*)alloc((size_t)4 * Ee * Hh * 2);
  __bf16* wpr_bf = (__bf16*)alloc((size_t)Ee * 4 * Ee * 2);
  __bf16* wq_bf  = (__bf16*)alloc((size_t)Mm * Hh * 2);
  __bf16* wk_bf  = (__bf16*)alloc((size_t)Mm * Hh * 2);
  __bf16* emb_bf = (__bf16*)alloc((size_t)Vv * Ee * 2);

  // Weight / embedding casts (bf16 operands for WMMA; all L2-resident).
  cast_bf16<<<512, 256, 0, stream>>>(w_ih, wih_bf, (long)3 * Hh * Ee);
  cast_bf16<<<512, 256, 0, stream>>>(w_fc, wfc_bf, (long)4 * Ee * Hh);
  cast_bf16<<<512, 256, 0, stream>>>(w_pr, wpr_bf, (long)Ee * 4 * Ee);
  cast_bf16<<<128, 256, 0, stream>>>(wq, wq_bf, (long)Mm * Hh);
  cast_bf16<<<128, 256, 0, stream>>>(wk, wk_bf, (long)Mm * Hh);
  cast_bf16<<<2048, 256, 0, stream>>>(emb, emb_bf, (long)Vv * Ee);
  gather_embed<<<R, 256, 0, stream>>>(ids, emb, x_bf);
  zero_f32<<<(2 * Bb * Hh + 255) / 256, 256, 0, stream>>>(hbuf, 2 * Bb * Hh);
  zero_u32<<<1, 32, 0, stream>>>(bar, 1);

  // xg = x @ w_ih^T + b_ih          (2048 x 3072, K=512)
  wmma_gemm_bias<<<dim3(3 * Hh / 128, R / 64), 256, 0, stream>>>(
      x_bf, wih_bf, b_ih, xg, nullptr, R, 3 * Hh, Ee, 0);
  // sequential recurrence -> states
  gru_scan<<<Hh / 16, 256, 0, stream>>>(xg, w_hh, b_hh, states, hbuf, bar);
  // layernorm + gate
  ln_gate<<<R, 256, 0, stream>>>(states, ln_g, ln_b, wg, bg, ns_bf, st_bf, gate);
  // q / k projections                (2048 x 128, K=1024)
  wmma_gemm_bias<<<dim3(Mm / 128, R / 64), 256, 0, stream>>>(
      st_bf, wq_bf, bq, qb, nullptr, R, Mm, Hh, 0);
  wmma_gemm_bias<<<dim3(Mm / 128, R / 64), 256, 0, stream>>>(
      st_bf, wk_bf, bk, kb, nullptr, R, Mm, Hh, 0);
  // hf = relu(ns @ w_fc^T + b_fc)^2  (2048 x 2048, K=1024), fused epilogue
  wmma_gemm_bias<<<dim3(4 * Ee / 128, R / 64), 256, 0, stream>>>(
      ns_bf, wfc_bf, b_fc, hf, hf_bf, R, 4 * Ee, Hh, 1);
  // base = hf @ w_pr^T + b_pr        (2048 x 512, K=2048)
  wmma_gemm_bias<<<dim3(Ee / 128, R / 64), 256, 0, stream>>>(
      hf_bf, wpr_bf, b_pr, bs, bs_bf, R, Ee, 4 * Ee, 0);
  // logits = base @ emb^T + out_bias (2048 x 32000, K=512) -- dominant GEMM
  wmma_gemm_bias<<<dim3(Vv / 128, R / 64), 256, 0, stream>>>(
      bs_bf, emb_bf, outb, out, nullptr, R, Vv, Ee, 0);
  // attention + scatter-add into logits
  attn_scatter<<<R, 256, 0, stream>>>(qb, kb, gate, ids, msc, out);
}
